// GraphPooling_13821204758763
// MI455X (gfx1250) — matrix operand order repaired and checked by reading, not compiled, not run on hardware
//
#include <hip/hip_runtime.h>

#define HDIM 256
#define ODIM 128
#define GNUM 16384
#define NROWS 500000
#define NSTRIPS (NROWS / 16)   // 31250, exact
#define PITCH 264              // bf16 elems per staged LDS row (pad vs 256 to rotate banks)

typedef __bf16 bf16_t;
typedef __attribute__((ext_vector_type(16))) __bf16 v16bf;
typedef __attribute__((ext_vector_type(8)))  __bf16 v8bf;
typedef __attribute__((ext_vector_type(8)))  float  v8f;

union V16U { v16bf v; v8bf h[2]; };
__device__ __forceinline__ v16bf mk16(v8bf lo, v8bf hi) { V16U u; u.h[0] = lo; u.h[1] = hi; return u.v; }
__device__ __forceinline__ bf16_t f2bf(float f) { return (bf16_t)f; }
__device__ __forceinline__ v8f zero8() { v8f z = {}; return z; }

// CDNA5 async global->LDS copy, 16B per lane, tracked by ASYNCcnt (no VGPR round-trip)
__device__ __forceinline__ void async_ld16(unsigned lds_off, const void* gsrc) {
    asm volatile("global_load_async_to_lds_b128 %0, %1, off"
                 :: "v"(lds_off), "v"(gsrc)
                 : "memory");
}
__device__ __forceinline__ void wait_async0() {
    asm volatile("s_wait_asynccnt 0x0" ::: "memory");
}

// ---------------- zero workspace (sums + counts are contiguous) ----------------
__global__ void zero_kernel(float* __restrict__ p, int n) {
    int i = blockIdx.x * blockDim.x + threadIdx.x;
    if (i < n) p[i] = 0.f;
}

// ---------------- pack fp32 weights into bf16 WMMA B-fragment layout ----------------
// Layout: frag f = kt*ntiles + nt, each frag = [plane p(2)][lane l(32)][j(8)] bf16
//   element = W[k][n], k = 32*kt + (l>>4)*8 + p*16 + j, n = 16*nt + (l&15)
__global__ void pack_w(const float* __restrict__ W, bf16_t* __restrict__ Wp,
                       int ntiles, int ncols) {
    int t = blockIdx.x * blockDim.x + threadIdx.x;
    int total = 8 * ntiles * 512;
    if (t >= total) return;
    int j  = t & 7;
    int l  = (t >> 3) & 31;
    int p  = (t >> 8) & 1;
    int f  = t >> 9;
    int kt = f / ntiles;
    int nt = f - kt * ntiles;
    int k  = kt * 32 + ((l >> 4) << 3) + p * 16 + j;
    int n  = nt * 16 + (l & 15);
    Wp[t] = f2bf(W[k * ncols + n]);
}

// ---------------- fused MLP + segment-sum ----------------
__launch_bounds__(256, 1)
__global__ void mlp_seg_kernel(const float* __restrict__ x,
                               const int*   __restrict__ batch,
                               const float* __restrict__ b1,
                               const float* __restrict__ b2,
                               const bf16_t* __restrict__ W1p,
                               const bf16_t* __restrict__ W2p,
                               float* __restrict__ sums,
                               float* __restrict__ counts) {
    __shared__ __align__(16) bf16_t wlds[8 * 16 * 512];    // 128 KB: one weight's fragments
    __shared__ __align__(16) bf16_t xlds[8 * 16 * PITCH];  // ~66 KB: per-wave x/h strip

    const int tid   = threadIdx.x;
    const int lane  = tid & 31;
    const int wave  = tid >> 5;
    const int arow  = lane & 15;
    const int mhalf = lane >> 4;

    const unsigned wlds_base = (unsigned)(uintptr_t)(&wlds[0]);

    // async-stage W1 fragments into LDS (128 KB, 16B per lane per op)
    {
        const char* src = (const char*)W1p;
        for (int i = tid; i < 8192; i += 256)
            async_ld16(wlds_base + i * 16, src + (size_t)i * 16);
        wait_async0();
    }
    __syncthreads();

    const int  strip  = blockIdx.x * 8 + wave;
    const bool active = strip < NSTRIPS;
    const int  r0     = strip * 16;
    bf16_t* xw = xlds + wave * 16 * PITCH;

    v8f acc[16];
#pragma unroll
    for (int nt = 0; nt < 16; ++nt) acc[nt] = zero8();

    const int aoff = arow * PITCH + mhalf * 8;

    if (active) {
        // stage x strip (16 x 256 fp32 -> bf16), fully coalesced global reads
#pragma unroll 4
        for (int r = 0; r < 16; ++r) {
            const float4* rp = (const float4*)(x + (size_t)(r0 + r) * HDIM + lane * 8);
            float4 f0 = rp[0];
            float4 f1 = rp[1];
            v8bf hb;
            hb[0] = f2bf(f0.x); hb[1] = f2bf(f0.y); hb[2] = f2bf(f0.z); hb[3] = f2bf(f0.w);
            hb[4] = f2bf(f1.x); hb[5] = f2bf(f1.y); hb[6] = f2bf(f1.z); hb[7] = f2bf(f1.w);
            *(v8bf*)(xw + r * PITCH + lane * 8) = hb;
        }
        // GEMM1: h = x @ W1  (A and B fragment loads software-pipelined 1 tile ahead)
        v8bf alo = *(const v8bf*)(xw + aoff);
        v8bf ahi = *(const v8bf*)(xw + aoff + 16);
#pragma unroll
        for (int kt = 0; kt < 8; ++kt) {
            v16bf a = mk16(alo, ahi);
            if (kt < 7) {
                alo = *(const v8bf*)(xw + aoff + (kt + 1) * 32);
                ahi = *(const v8bf*)(xw + aoff + (kt + 1) * 32 + 16);
            }
            const bf16_t* fbase = wlds + kt * 16 * 512 + lane * 8;
            v8bf blo = *(const v8bf*)(fbase);
            v8bf bhi = *(const v8bf*)(fbase + 256);
#pragma unroll
            for (int nt = 0; nt < 16; ++nt) {
                v16bf b = mk16(blo, bhi);
                if (nt < 15) {
                    blo = *(const v8bf*)(fbase + (nt + 1) * 512);
                    bhi = *(const v8bf*)(fbase + (nt + 1) * 512 + 256);
                }
                acc[nt] = __builtin_amdgcn_wmma_f32_16x16x32_bf16(
                    false, a, false, b, (short)0, acc[nt], false, false);
            }
        }
        // bias + shifted softplus, write h back into the (now consumed) x strip
#pragma unroll
        for (int nt = 0; nt < 16; ++nt) {
            const int n = nt * 16 + arow;
            const float bias = b1[n];
#pragma unroll
            for (int v = 0; v < 8; ++v) {
                float t  = acc[nt][v] + bias;
                float sp = (t > 20.f) ? t : __logf(1.f + __expf(t));
                sp -= 0.69314718055994531f;  // shifted softplus
                xw[(v + mhalf * 8) * PITCH + n] = f2bf(sp);
            }
        }
    }

    __syncthreads();
    // async-overwrite weight LDS with W2 fragments
    {
        const char* src = (const char*)W2p;
        for (int i = tid; i < 8192; i += 256)
            async_ld16(wlds_base + i * 16, src + (size_t)i * 16);
        wait_async0();
    }
    __syncthreads();

    if (!active) return;

#pragma unroll
    for (int nt = 0; nt < 16; ++nt) acc[nt] = zero8();
    // GEMM2: h2 = h @ W2 (same pipelining)
    {
        v8bf alo = *(const v8bf*)(xw + aoff);
        v8bf ahi = *(const v8bf*)(xw + aoff + 16);
#pragma unroll
        for (int kt = 0; kt < 8; ++kt) {
            v16bf a = mk16(alo, ahi);
            if (kt < 7) {
                alo = *(const v8bf*)(xw + aoff + (kt + 1) * 32);
                ahi = *(const v8bf*)(xw + aoff + (kt + 1) * 32 + 16);
            }
            const bf16_t* fbase = wlds + kt * 16 * 512 + lane * 8;
            v8bf blo = *(const v8bf*)(fbase);
            v8bf bhi = *(const v8bf*)(fbase + 256);
#pragma unroll
            for (int nt = 0; nt < 16; ++nt) {
                v16bf b = mk16(blo, bhi);
                if (nt < 15) {
                    blo = *(const v8bf*)(fbase + (nt + 1) * 512);
                    bhi = *(const v8bf*)(fbase + (nt + 1) * 512 + 256);
                }
                acc[nt] = __builtin_amdgcn_wmma_f32_16x16x32_bf16(
                    false, a, false, b, (short)0, acc[nt], false, false);
            }
        }
    }

    // segment reduction; sorted batch -> usually one graph per 16-row strip
    const int gfirst = batch[r0];
    const int glast  = batch[r0 + 15];
    if (gfirst == glast) {
#pragma unroll
        for (int nt = 0; nt < 16; ++nt) {
            const int n = nt * 16 + arow;
            float s = 8.f * b2[n];   // bias for 8 rows of this half
#pragma unroll
            for (int v = 0; v < 8; ++v) s += acc[nt][v];
            s += __shfl_xor(s, 16, 32);          // combine both 8-row halves
            if (lane < 16) atomicAdd(&sums[(size_t)gfirst * HDIM + n], s);
        }
        if (lane == 0) atomicAdd(&counts[gfirst], 16.f);
    } else {
#pragma unroll
        for (int nt = 0; nt < 16; ++nt) {
            const int n = nt * 16 + arow;
            const float bias = b2[n];
#pragma unroll
            for (int v = 0; v < 8; ++v) {
                const int g = batch[r0 + v + mhalf * 8];
                atomicAdd(&sums[(size_t)g * HDIM + n], acc[nt][v] + bias);
            }
        }
        if (lane == 0) {
            for (int m = 0; m < 16; ++m) atomicAdd(&counts[batch[r0 + m]], 1.f);
        }
    }
}

// ---------------- mean + output GEMM: out = (sums/counts) @ Wo + bo ----------------
__launch_bounds__(256, 1)
__global__ void pool_out_kernel(const float* __restrict__ sums,
                                const float* __restrict__ counts,
                                const bf16_t* __restrict__ Wop,
                                const float* __restrict__ bo,
                                float* __restrict__ out) {
    const int tid   = threadIdx.x;
    const int lane  = tid & 31;
    const int wave  = tid >> 5;
    const int arow  = lane & 15;
    const int mhalf = lane >> 4;
    const int strip = blockIdx.x * 8 + wave;   // 1024 strips exactly
    const int g0    = strip * 16;

    const float inv = 1.f / fmaxf(counts[g0 + arow], 1.f);

    v8f acc[8];
#pragma unroll
    for (int nt = 0; nt < 8; ++nt) acc[nt] = zero8();

#pragma unroll
    for (int kt = 0; kt < 8; ++kt) {
        const int k0 = kt * 32 + mhalf * 8;
        const float4* rp = (const float4*)(sums + (size_t)(g0 + arow) * HDIM + k0);
        float4 f0 = rp[0], f1 = rp[1];
        float4 f2 = rp[4], f3 = rp[5];   // +16 floats
        v8bf alo, ahi;
        alo[0] = f2bf(f0.x * inv); alo[1] = f2bf(f0.y * inv);
        alo[2] = f2bf(f0.z * inv); alo[3] = f2bf(f0.w * inv);
        alo[4] = f2bf(f1.x * inv); alo[5] = f2bf(f1.y * inv);
        alo[6] = f2bf(f1.z * inv); alo[7] = f2bf(f1.w * inv);
        ahi[0] = f2bf(f2.x * inv); ahi[1] = f2bf(f2.y * inv);
        ahi[2] = f2bf(f2.z * inv); ahi[3] = f2bf(f2.w * inv);
        ahi[4] = f2bf(f3.x * inv); ahi[5] = f2bf(f3.y * inv);
        ahi[6] = f2bf(f3.z * inv); ahi[7] = f2bf(f3.w * inv);
        v16bf a = mk16(alo, ahi);

        const bf16_t* fbase = Wop + kt * 8 * 512 + lane * 8;
        v8bf blo = *(const v8bf*)(fbase);
        v8bf bhi = *(const v8bf*)(fbase + 256);
#pragma unroll
        for (int nt = 0; nt < 8; ++nt) {
            v16bf b = mk16(blo, bhi);
            if (nt < 7) {
                blo = *(const v8bf*)(fbase + (nt + 1) * 512);
                bhi = *(const v8bf*)(fbase + (nt + 1) * 512 + 256);
            }
            acc[nt] = __builtin_amdgcn_wmma_f32_16x16x32_bf16(
                false, a, false, b, (short)0, acc[nt], false, false);
        }
    }

#pragma unroll
    for (int nt = 0; nt < 8; ++nt) {
        const int n = nt * 16 + arow;
        const float bias = bo[n];
#pragma unroll
        for (int v = 0; v < 8; ++v) {
            out[(size_t)(g0 + v + mhalf * 8) * ODIM + n] = acc[nt][v] + bias;
        }
    }
}

// ---------------- host launcher ----------------
extern "C" void kernel_launch(void* const* d_in, const int* in_sizes, int n_in,
                              void* d_out, int out_size, void* d_ws, size_t ws_size,
                              hipStream_t stream) {
    const float* x     = (const float*)d_in[0];
    const int*   batch = (const int*)d_in[1];
    // d_in[2] = num_graphs (compile-time constant GNUM)
    const float* W1 = (const float*)d_in[3];
    const float* b1 = (const float*)d_in[4];
    const float* W2 = (const float*)d_in[5];
    const float* b2 = (const float*)d_in[6];
    const float* Wo = (const float*)d_in[7];
    const float* bo = (const float*)d_in[8];
    float* out = (float*)d_out;

    char* ws = (char*)d_ws;
    float*  sums   = (float*)ws;                                        // G*H f32 = 16 MB
    float*  counts = (float*)(ws + (size_t)GNUM * HDIM * 4);            // G f32
    bf16_t* W1p    = (bf16_t*)(ws + (size_t)GNUM * HDIM * 4 + GNUM * 4);
    bf16_t* W2p    = W1p + 8 * 16 * 512;
    bf16_t* Wop    = W2p + 8 * 16 * 512;

    const int nzero = GNUM * HDIM + GNUM;
    zero_kernel<<<(nzero + 255) / 256, 256, 0, stream>>>(sums, nzero);

    pack_w<<<(8 * 16 * 512) / 256, 256, 0, stream>>>(W1, W1p, 16, HDIM);
    pack_w<<<(8 * 16 * 512) / 256, 256, 0, stream>>>(W2, W2p, 16, HDIM);
    pack_w<<<(8 * 8 * 512) / 256, 256, 0, stream>>>(Wo, Wop, 8, ODIM);

    mlp_seg_kernel<<<(NSTRIPS + 7) / 8, 256, 0, stream>>>(
        x, batch, b1, b2, W1p, W2p, sums, counts);

    pool_out_kernel<<<(GNUM / 16) / 8, 256, 0, stream>>>(sums, counts, Wop, bo, out);
}
